// GroupedMLP_3530463117599
// MI455X (gfx1250) — matrix operand order repaired and checked by reading, not compiled, not run on hardware
//
#include <hip/hip_runtime.h>
#include <hip/hip_bf16.h>

typedef __attribute__((ext_vector_type(16))) __bf16 v16bf;
typedef __attribute__((ext_vector_type(8)))  float  v8f;
typedef unsigned int v4u __attribute__((ext_vector_type(4)));
typedef int          v8i __attribute__((ext_vector_type(8)));
typedef int          v4i __attribute__((ext_vector_type(4)));

#define T_TOK 16384
#define HDIM  2048
#define IDIM  1408
#define NEXP  8
#define N2    (2 * IDIM)   // 2816

#define LSTR  40           // LDS row stride (bf16 elems): 32 data + 8 pad

union FragB { v16bf bf; unsigned int u[8]; };

__device__ __forceinline__ unsigned short f2bf(float f) {
  unsigned int u = __float_as_uint(f);
  u += 0x7FFFu + ((u >> 16) & 1u);   // round-to-nearest-even
  return (unsigned short)(u >> 16);
}

// ------------------------------------------------------------------ TDM tile
// 2D bf16 tile load (tile_d0 elems wide x tile_d1 rows) global -> LDS via the
// Tensor Data Mover, with LDS padding so each 32-elem (64B = 16 DWORD) row is
// followed by 16B of pad -> matches the LSTR=40 layout.
//   pad_interval = 3 -> pad every 16 DWORDs; pad_amount = 3 -> 4 DWORDs pad.
// OOB rows/cols (beyond tensor_d0/tensor_d1 from the tile origin) read as 0.
__device__ __forceinline__ void tdm_load_tile(
    unsigned lds_byte_addr, const void* gaddr,
    unsigned tensor_d0, unsigned tensor_d1, unsigned tensor_d0_stride,
    unsigned tile_d0, unsigned tile_d1) {
  unsigned long long ga = (unsigned long long)(size_t)gaddr;
  v4u g0;
  g0[0] = 1u;                                            // count=1 (valid), no gather
  g0[1] = lds_byte_addr;                                 // lds_addr
  g0[2] = (unsigned)ga;                                  // global_addr[31:0]
  g0[3] = (unsigned)((ga >> 32) & 0x01FFFFFFull) | (2u << 30); // addr[56:32], type=2
  v8i g1;
  g1[0] = (int)((1u << 16)      // data_size = 1 (2 bytes)
              | (1u << 20)      // pad_enable
              | (3u << 22)      // pad_interval: 16 DWORDs
              | (3u << 25));    // pad_amount: 4 DWORDs
  g1[1] = (int)((tensor_d0 & 0xFFFFu) << 16);                               // tensor_dim0[15:0]
  g1[2] = (int)(((tensor_d0 >> 16) & 0xFFFFu) | ((tensor_d1 & 0xFFFFu) << 16));
  g1[3] = (int)(((tensor_d1 >> 16) & 0xFFFFu) | ((tile_d0 & 0xFFFFu) << 16)); // tile_dim0
  g1[4] = (int)(tile_d1 & 0xFFFFu);                                         // tile_dim1 (dim2=0)
  g1[5] = (int)tensor_d0_stride;                                            // dim0 stride[31:0]
  g1[6] = 0;
  g1[7] = 0;
  v4i z4 = {0, 0, 0, 0};
#if defined(__clang_major__) && (__clang_major__ >= 23)
  v8i z8 = {0, 0, 0, 0, 0, 0, 0, 0};
  __builtin_amdgcn_tensor_load_to_lds(g0, g1, z4, z4, z8, 0);
#else
  __builtin_amdgcn_tensor_load_to_lds(g0, g1, z4, z4, 0);
#endif
}

// ---------------------------------------------------------------- fp32 -> bf16
__global__ __launch_bounds__(256) void cvt_bf16_kernel(
    const float* __restrict__ src, unsigned short* __restrict__ dst, size_t n8) {
  size_t i = (size_t)blockIdx.x * blockDim.x + threadIdx.x;
  const size_t stride = (size_t)gridDim.x * blockDim.x;
  for (; i < n8; i += stride) {
    const float4* p = (const float4*)(src + i * 8);
    float4 a = p[0], b = p[1];
    uint4 o;
    o.x = (unsigned)f2bf(a.x) | ((unsigned)f2bf(a.y) << 16);
    o.y = (unsigned)f2bf(a.z) | ((unsigned)f2bf(a.w) << 16);
    o.z = (unsigned)f2bf(b.x) | ((unsigned)f2bf(b.y) << 16);
    o.w = (unsigned)f2bf(b.z) | ((unsigned)f2bf(b.w) << 16);
    ((uint4*)dst)[i] = o;
  }
}

// ---------------------------------------------------------------- prefix sums
__global__ void prefix_kernel(const int* __restrict__ counts, int* __restrict__ off) {
  if (threadIdx.x == 0 && blockIdx.x == 0) {
    int s = 0;
    off[0] = 0;
    for (int e = 0; e < NEXP; ++e) { s += counts[e]; off[e + 1] = s; }
  }
}

// ---------------------------------------------------------------- GEMM1 + SwiGLU
// x[T,H](bf16) @ Wgu[E,H,2I](bf16) -> silu(gate)*up -> h[T,I](bf16)
// grid: (I/64, E, T/128), block 256 (8 waves: 4 along M x 2 along N)
__global__ __launch_bounds__(256) void gemm1_swiglu_kernel(
    const unsigned short* __restrict__ X,
    const unsigned short* __restrict__ W,
    const int* __restrict__ off,
    unsigned short* __restrict__ Hout) {
  __shared__ alignas(16) unsigned short sA[128 * LSTR];
  __shared__ alignas(16) unsigned short sBg[64 * LSTR];   // transposed [n][k]
  __shared__ alignas(16) unsigned short sBu[64 * LSTR];   // transposed [n][k]

  const int e       = blockIdx.y;
  const int rowBase = off[e] + blockIdx.z * 128;
  const int rowEnd  = off[e + 1];
  if (rowBase >= rowEnd) return;
  const int n0 = blockIdx.x * 64;

  const int t     = threadIdx.x;
  const int lane  = t & 31;
  const int wave  = t >> 5;
  const int waveM = wave & 3;
  const int waveN = wave >> 2;
  const int lrow  = lane & 15;
  const int khalf = lane >> 4;

  // A-tile: each wave DMAs its own 16-row slice via TDM
  const int waveRow = wave * 16;
  const int gRow0   = rowBase + waveRow;
  int remRows = rowEnd - gRow0;
  if (remRows < 0) remRows = 0;
  const unsigned ldsA = (unsigned)(size_t)(&sA[waveRow * LSTR]);

  const int bk = t >> 3;            // 0..31
  const int bn = (t & 7) * 8;       // 0..56
  const size_t wBaseG = (size_t)e * HDIM * N2 + (size_t)(n0 + bn);
  const size_t wBaseU = wBaseG + IDIM;

  const v8f vzero = {0.f, 0.f, 0.f, 0.f, 0.f, 0.f, 0.f, 0.f};
  v8f accg[2][2], accu[2][2];
#pragma unroll
  for (int m = 0; m < 2; ++m)
#pragma unroll
    for (int n = 0; n < 2; ++n) { accg[m][n] = vzero; accu[m][n] = vzero; }

  for (int k0 = 0; k0 < HDIM; k0 += 32) {
    uint4 gv = *(const uint4*)(W + wBaseG + (size_t)(k0 + bk) * N2);
    uint4 uv = *(const uint4*)(W + wBaseU + (size_t)(k0 + bk) * N2);

    __syncthreads();   // previous iteration done reading LDS
    tdm_load_tile(ldsA, X + (size_t)gRow0 * HDIM + k0,
                  (unsigned)(HDIM - k0), (unsigned)remRows, HDIM, 32, 16);
    {
      union { uint4 v; unsigned short s[8]; } gg, uu;
      gg.v = gv; uu.v = uv;
#pragma unroll
      for (int j = 0; j < 8; ++j) {
        sBg[(bn + j) * LSTR + bk] = gg.s[j];
        sBu[(bn + j) * LSTR + bk] = uu.s[j];
      }
    }
    __builtin_amdgcn_s_wait_tensorcnt(0);
    __syncthreads();

    FragB a[2], bg[2], bu[2];
#pragma unroll
    for (int m = 0; m < 2; ++m) {
      const int r = (waveM * 32 + m * 16 + lrow) * LSTR;
#pragma unroll
      for (int v = 0; v < 8; ++v) {
        const int kk = khalf * 16 + 2 * v;
        a[m].u[v] = *(const unsigned int*)&sA[r + kk];
      }
    }
    const int koff = khalf * 16;
#pragma unroll
    for (int n = 0; n < 2; ++n) {
      const int c = (waveN * 32 + n * 16 + lrow) * LSTR + koff;
#pragma unroll
      for (int v = 0; v < 8; ++v) {
        bg[n].u[v] = *(const unsigned int*)&sBg[c + 2 * v];
        bu[n].u[v] = *(const unsigned int*)&sBu[c + 2 * v];
      }
    }
#pragma unroll
    for (int m = 0; m < 2; ++m)
#pragma unroll
      for (int n = 0; n < 2; ++n) {
        accg[m][n] = __builtin_amdgcn_wmma_f32_16x16x32_bf16(
            false, a[m].bf, false, bg[n].bf, (short)0, accg[m][n], false, false);
        accu[m][n] = __builtin_amdgcn_wmma_f32_16x16x32_bf16(
            false, a[m].bf, false, bu[n].bf, (short)0, accu[m][n], false, false);
      }
  }

  // epilogue: h = silu(gate) * up, bf16 store (fast rcp instead of IEEE divide)
#pragma unroll
  for (int m = 0; m < 2; ++m) {
#pragma unroll
    for (int n = 0; n < 2; ++n) {
      const int col = n0 + waveN * 32 + n * 16 + lrow;
#pragma unroll
      for (int r = 0; r < 8; ++r) {
        const int row = rowBase + waveM * 32 + m * 16 + khalf * 8 + r;
        if (row < rowEnd) {
          const float g = accg[m][n][r];
          const float u = accu[m][n][r];
          const float s = __builtin_amdgcn_rcpf(1.0f + __expf(-g));
          Hout[(size_t)row * IDIM + col] = f2bf(g * s * u);
        }
      }
    }
  }
}

// ---------------------------------------------------------------- GEMM2
// h[T,I](bf16) @ Wd[E,I,H](bf16) -> out[T,H](fp32)
// grid: (H/128, E, T/128), block 256 (8 waves: 4 along M x 2 along N, wave 32x64)
__global__ __launch_bounds__(256) void gemm2_kernel(
    const unsigned short* __restrict__ Hin,
    const unsigned short* __restrict__ W,
    const int* __restrict__ off,
    float* __restrict__ Out) {
  __shared__ alignas(16) unsigned short sA[128 * LSTR];
  __shared__ alignas(16) unsigned short sB[128 * LSTR];   // transposed [n][k]

  const int e       = blockIdx.y;
  const int rowBase = off[e] + blockIdx.z * 128;
  const int rowEnd  = off[e + 1];
  if (rowBase >= rowEnd) return;
  const int n0 = blockIdx.x * 128;

  const int t     = threadIdx.x;
  const int lane  = t & 31;
  const int wave  = t >> 5;
  const int waveM = wave & 3;
  const int waveN = wave >> 2;
  const int lrow  = lane & 15;
  const int khalf = lane >> 4;

  const int waveRow = wave * 16;
  const int gRow0   = rowBase + waveRow;
  int remRows = rowEnd - gRow0;
  if (remRows < 0) remRows = 0;
  const unsigned ldsA = (unsigned)(size_t)(&sA[waveRow * LSTR]);

  const int bk = t >> 3;            // 0..31
  const int bn = (t & 7) * 8;       // 0..56, two 64-col halves
  const size_t wBase = (size_t)e * IDIM * HDIM + (size_t)(n0 + bn);

  const v8f vzero = {0.f, 0.f, 0.f, 0.f, 0.f, 0.f, 0.f, 0.f};
  v8f acc[2][4];
#pragma unroll
  for (int m = 0; m < 2; ++m)
#pragma unroll
    for (int n = 0; n < 4; ++n) acc[m][n] = vzero;

  for (int k0 = 0; k0 < IDIM; k0 += 32) {
    uint4 bv0 = *(const uint4*)(W + wBase + (size_t)(k0 + bk) * HDIM);
    uint4 bv1 = *(const uint4*)(W + wBase + (size_t)(k0 + bk) * HDIM + 64);

    __syncthreads();
    tdm_load_tile(ldsA, Hin + (size_t)gRow0 * IDIM + k0,
                  (unsigned)(IDIM - k0), (unsigned)remRows, IDIM, 32, 16);
    {
      union { uint4 v; unsigned short s[8]; } b0, b1;
      b0.v = bv0; b1.v = bv1;
#pragma unroll
      for (int j = 0; j < 8; ++j) {
        sB[(bn + j) * LSTR + bk]      = b0.s[j];
        sB[(bn + 64 + j) * LSTR + bk] = b1.s[j];
      }
    }
    __builtin_amdgcn_s_wait_tensorcnt(0);
    __syncthreads();

    FragB a[2], b[4];
#pragma unroll
    for (int m = 0; m < 2; ++m) {
      const int r = (waveM * 32 + m * 16 + lrow) * LSTR;
#pragma unroll
      for (int v = 0; v < 8; ++v) {
        const int kk = khalf * 16 + 2 * v;
        a[m].u[v] = *(const unsigned int*)&sA[r + kk];
      }
    }
    const int koff = khalf * 16;
#pragma unroll
    for (int n = 0; n < 4; ++n) {
      const int c = (waveN * 64 + n * 16 + lrow) * LSTR + koff;
#pragma unroll
      for (int v = 0; v < 8; ++v)
        b[n].u[v] = *(const unsigned int*)&sB[c + 2 * v];
    }
#pragma unroll
    for (int m = 0; m < 2; ++m)
#pragma unroll
      for (int n = 0; n < 4; ++n)
        acc[m][n] = __builtin_amdgcn_wmma_f32_16x16x32_bf16(
            false, a[m].bf, false, b[n].bf, (short)0, acc[m][n], false, false);
  }

#pragma unroll
  for (int m = 0; m < 2; ++m) {
#pragma unroll
    for (int n = 0; n < 4; ++n) {
      const int col = n0 + waveN * 64 + n * 16 + lrow;
#pragma unroll
      for (int r = 0; r < 8; ++r) {
        const int row = rowBase + waveM * 32 + m * 16 + khalf * 8 + r;
        if (row < rowEnd) Out[(size_t)row * HDIM + col] = acc[m][n][r];
      }
    }
  }
}

// ---------------------------------------------------------------- launcher
extern "C" void kernel_launch(void* const* d_in, const int* in_sizes, int n_in,
                              void* d_out, int out_size, void* d_ws, size_t ws_size,
                              hipStream_t stream) {
  (void)in_sizes; (void)n_in; (void)out_size; (void)ws_size;
  const float* hs     = (const float*)d_in[0];   // [T, H]
  const float* wgu    = (const float*)d_in[1];   // [E, H, 2I]
  const float* wd     = (const float*)d_in[2];   // [E, I, H]
  const int*   counts = (const int*)d_in[3];     // [E]
  float*       out    = (float*)d_out;           // [T, H]

  char* ws = (char*)d_ws;
  int*            off   = (int*)ws;                                    // [E+1]
  unsigned short* Xbf   = (unsigned short*)(ws + 256);                 // [T,H]
  unsigned short* Wgubf = Xbf   + (size_t)T_TOK * HDIM;                // [E,H,2I]
  unsigned short* Wdbf  = Wgubf + (size_t)NEXP * HDIM * N2;            // [E,I,H]
  unsigned short* Hbf   = Wdbf  + (size_t)NEXP * IDIM * HDIM;          // [T,I]

  cvt_bf16_kernel<<<4096, 256, 0, stream>>>(hs,  Xbf,   (size_t)T_TOK * HDIM / 8);
  cvt_bf16_kernel<<<8192, 256, 0, stream>>>(wgu, Wgubf, (size_t)NEXP * HDIM * N2 / 8);
  cvt_bf16_kernel<<<4096, 256, 0, stream>>>(wd,  Wdbf,  (size_t)NEXP * IDIM * HDIM / 8);
  prefix_kernel<<<1, 1, 0, stream>>>(counts, off);

  dim3 g1(IDIM / 64, NEXP, T_TOK / 128);
  gemm1_swiglu_kernel<<<g1, 256, 0, stream>>>(Xbf, Wgubf, off, Hbf);

  dim3 g2(HDIM / 128, NEXP, T_TOK / 128);
  gemm2_kernel<<<g2, 256, 0, stream>>>(Hbf, Wdbf, off, out);
}
